// AggregationFusion_77884936946227
// MI455X (gfx1250) — matrix operand order repaired and checked by reading, not compiled, not run on hardware
//
#include <hip/hip_runtime.h>
#include <stdint.h>

typedef unsigned short u16;
typedef __attribute__((ext_vector_type(16))) __bf16 v16bf;
typedef __attribute__((ext_vector_type(8)))  float  v8f;
typedef __attribute__((ext_vector_type(4)))  int    vi4;
typedef __attribute__((ext_vector_type(4)))  unsigned int vu4;
typedef __attribute__((ext_vector_type(4)))  float  vf4;

#define FDIM   512
#define TWOF   1024
#define MT     64            // rows per workgroup
#define ROWB   2064          // LDS row stride in bytes: 1024*2 + 16B pad (bank spread)
#define LN_EPS 1e-5f

union FragU { vi4 q[2]; v16bf v; };

__device__ __forceinline__ u16 f2bf(float f) {
  uint32_t u = __float_as_uint(f);
  u += 0x7FFFu + ((u >> 16) & 1u);           // round-to-nearest-even
  return (u16)(u >> 16);
}
__device__ __forceinline__ float bf2f(u16 h) {
  return __uint_as_float(((uint32_t)h) << 16);
}
__device__ __forceinline__ float wave_sum(float v) {
  v += __shfl_xor(v, 16, 32);
  v += __shfl_xor(v,  8, 32);
  v += __shfl_xor(v,  4, 32);
  v += __shfl_xor(v,  2, 32);
  v += __shfl_xor(v,  1, 32);
  return v;
}
__device__ __forceinline__ float silu(float x) { return x / (1.0f + __expf(-x)); }

// A fragment (16x32 bf16) from LDS. ISA layout: lane m (0-15) holds row m,
// K = half*8 + [0..8) in v0-3 and half*8 + 16 + [0..8) in v4-7  (half = lane>=16)
__device__ __forceinline__ v16bf lds_a_frag(const char* base, int m0, int k0, int lane) {
  const int half = lane >> 4, m = lane & 15;
  const char* p = base + (size_t)(m0 + m) * ROWB + (size_t)(k0 + half * 8) * 2;
  FragU u;
  u.q[0] = *(const vi4*)(p);
  u.q[1] = *(const vi4*)(p + 32);
  return u.v;
}
// B fragment (32x16 bf16) from global n-major bf16 weights (ldk = TWOF).
// Lane n (0-15) holds column n, K = half*8 + [0..8) in v0-3 and half*8 + 16 + [0..8) in v4-7.
__device__ __forceinline__ v16bf glb_b_frag(const u16* __restrict__ WT, int n0, int k0, int lane) {
  const int half = lane >> 4, n = lane & 15;
  const char* p = (const char*)(WT + (size_t)(n0 + n) * TWOF + (size_t)(k0 + half * 8));
  FragU u;
  u.q[0] = *(const vi4*)(p);
  u.q[1] = *(const vi4*)(p + 32);
  return u.v;
}
__device__ __forceinline__ v8f wmma_bf16(v16bf a, v16bf b, v8f c) {
  return __builtin_amdgcn_wmma_f32_16x16x32_bf16(false, a, false, b, (short)0, c,
                                                 false, false);
}

// ---------------------------------------------------------------------------
// Weight prep: transpose f32 (K x Nn, row-major) -> bf16 (Nn x K, row-major)
// ---------------------------------------------------------------------------
__global__ __launch_bounds__(256)
void transpose_f32_to_bf16(const float* __restrict__ in, u16* __restrict__ out,
                           int K, int Nn) {
  __shared__ float tile[32][33];
  const int bn = blockIdx.x * 32;   // column (n) tile
  const int bk = blockIdx.y * 32;   // row (k) tile
  const int tx = threadIdx.x;       // 0..31
  const int ty = threadIdx.y;       // 0..7
#pragma unroll
  for (int i = 0; i < 32; i += 8)
    tile[ty + i][tx] = in[(size_t)(bk + ty + i) * Nn + (bn + tx)];
  __syncthreads();
#pragma unroll
  for (int i = 0; i < 32; i += 8)
    out[(size_t)(bn + ty + i) * K + (bk + tx)] = f2bf(tile[tx][ty + i]);
}

// ---------------------------------------------------------------------------
// Fused: gather -> LN1 -> GEMM1(1024x1024)+SiLU -> LN2 -> GEMM2(1024x512)+SiLU
// One workgroup = 256 threads (8 waves), MT=64 rows, intermediates in LDS.
// ---------------------------------------------------------------------------
__global__ __launch_bounds__(256, 1)
void fused_mlp_kernel(const float* __restrict__ nodes,
                      const int*   __restrict__ comps,
                      const float* __restrict__ aggr_nodes,
                      const int*   __restrict__ aggr_comps,
                      const float* __restrict__ ln1_g, const float* __restrict__ ln1_b,
                      const u16*   __restrict__ W1T,   const float* __restrict__ b1,
                      const float* __restrict__ ln2_g, const float* __restrict__ ln2_b,
                      const u16*   __restrict__ W2T,   const float* __restrict__ b2,
                      float* __restrict__ out, int N, int M) {
  extern __shared__ char smem[];
  char* XN = smem;                        // MT x ROWB : LN1(combined) in bf16
  char* H1 = smem + (size_t)MT * ROWB;    // MT x ROWB : hidden in bf16

  const int tid  = threadIdx.x;
  const int wave = tid >> 5;
  const int lane = tid & 31;
  const int row0 = blockIdx.x * MT;

  // ---------------- Phase 1: gather + LayerNorm1 -> XN (bf16) ----------------
#pragma unroll 1
  for (int i = 0; i < 8; ++i) {
    const int rl = wave * 8 + i;            // local row, one row per wave pass
    const int rg = row0 + rl;
    const bool valid = (rg < N);
    const int key = valid ? comps[rg] : 0;
    // searchsorted-left over sorted aggr_comps (uniform across the wave)
    int lo = 0, hi = M;
    while (lo < hi) { int mid = (lo + hi) >> 1; if (aggr_comps[mid] < key) lo = mid + 1; else hi = mid; }
    const int idx = (lo < M) ? lo : (M - 1);

    const vf4* nrow = (const vf4*)(nodes      + (size_t)(valid ? rg : 0) * FDIM);
    const vf4* arow = (const vf4*)(aggr_nodes + (size_t)idx * FDIM);

    vf4 x[8];
    float s = 0.f, ss = 0.f;
#pragma unroll
    for (int j = 0; j < 8; ++j) {
      const int c4 = lane + 32 * j;         // float4 index 0..255 over 1024 cols
      vf4 v;
      if (!valid)       v = (vf4){0.f, 0.f, 0.f, 0.f};
      else if (j < 4)   v = nrow[c4];
      else              v = arow[c4 - 128];
      x[j] = v;
      s  += v.x + v.y + v.z + v.w;
      ss += v.x * v.x + v.y * v.y + v.z * v.z + v.w * v.w;
    }
    s  = wave_sum(s);
    ss = wave_sum(ss);
    const float mu = s * (1.0f / TWOF);
    const float var = ss * (1.0f / TWOF) - mu * mu;
    const float rs = rsqrtf(var + LN_EPS);
#pragma unroll
    for (int j = 0; j < 8; ++j) {
      const int c = (lane + 32 * j) * 4;
      const vf4 g = *(const vf4*)(ln1_g + c);
      const vf4 b = *(const vf4*)(ln1_b + c);
      const vf4 v = x[j];
      const u16 h0 = f2bf((v.x - mu) * rs * g.x + b.x);
      const u16 h1 = f2bf((v.y - mu) * rs * g.y + b.y);
      const u16 h2 = f2bf((v.z - mu) * rs * g.z + b.z);
      const u16 h3 = f2bf((v.w - mu) * rs * g.w + b.w);
      uint2 pk;
      pk.x = (uint32_t)h0 | ((uint32_t)h1 << 16);
      pk.y = (uint32_t)h2 | ((uint32_t)h3 << 16);
      *(uint2*)(XN + (size_t)rl * ROWB + (size_t)c * 2) = pk;
    }
  }
  __syncthreads();

  // ---------------- Phase 2: GEMM1 (64x1024 @ 1024x1024) + bias + SiLU -------
  // wave owns 128 N-columns; 2-tile N blocking reuses each A fragment.
#pragma unroll 1
  for (int mt = 0; mt < 4; ++mt) {
    const int m0 = mt * 16;
#pragma unroll 1
    for (int np = 0; np < 4; ++np) {
      const int n0 = wave * 128 + np * 32;
      v8f acc0 = {}, acc1 = {};
#pragma unroll 4
      for (int kt = 0; kt < 32; ++kt) {
        const int k0 = kt * 32;
        const v16bf a  = lds_a_frag(XN, m0, k0, lane);
        const v16bf bA = glb_b_frag(W1T, n0,      k0, lane);
        const v16bf bB = glb_b_frag(W1T, n0 + 16, k0, lane);
        acc0 = wmma_bf16(a, bA, acc0);
        acc1 = wmma_bf16(a, bB, acc1);
      }
      const int half = lane >> 4;
      const int nA = n0 + (lane & 15), nB = nA + 16;
      const float biA = b1[nA], biB = b1[nB];
#pragma unroll
      for (int r = 0; r < 8; ++r) {
        const int m = m0 + r + 8 * half;
        const float v0 = silu(acc0[r] + biA);
        const float v1 = silu(acc1[r] + biB);
        *(u16*)(H1 + (size_t)m * ROWB + (size_t)nA * 2) = f2bf(v0);
        *(u16*)(H1 + (size_t)m * ROWB + (size_t)nB * 2) = f2bf(v1);
      }
    }
  }
  __syncthreads();

  // ---------------- Phase 3: LayerNorm2 in-place on H1 -----------------------
#pragma unroll 1
  for (int i = 0; i < 8; ++i) {
    const int rl = wave * 8 + i;
    char* rowp = H1 + (size_t)rl * ROWB;
    vu4 raw[4];
    float s = 0.f, ss = 0.f;
#pragma unroll
    for (int j = 0; j < 4; ++j) {
      const int c = (lane + 32 * j) * 8;     // 8 bf16 per lane per j
      raw[j] = *(const vu4*)(rowp + (size_t)c * 2);
#pragma unroll
      for (int d = 0; d < 4; ++d) {
        const uint32_t w = raw[j][d];
        const float lo = bf2f((u16)(w & 0xFFFFu));
        const float hi = bf2f((u16)(w >> 16));
        s += lo + hi; ss += lo * lo + hi * hi;
      }
    }
    s  = wave_sum(s);
    ss = wave_sum(ss);
    const float mu = s * (1.0f / TWOF);
    const float var = ss * (1.0f / TWOF) - mu * mu;
    const float rs = rsqrtf(var + LN_EPS);
#pragma unroll
    for (int j = 0; j < 4; ++j) {
      const int c = (lane + 32 * j) * 8;
      const vf4 g0 = *(const vf4*)(ln2_g + c);
      const vf4 g1 = *(const vf4*)(ln2_g + c + 4);
      const vf4 b0 = *(const vf4*)(ln2_b + c);
      const vf4 b1v = *(const vf4*)(ln2_b + c + 4);
      float gv[8] = {g0.x, g0.y, g0.z, g0.w, g1.x, g1.y, g1.z, g1.w};
      float bv[8] = {b0.x, b0.y, b0.z, b0.w, b1v.x, b1v.y, b1v.z, b1v.w};
      vu4 o;
#pragma unroll
      for (int d = 0; d < 4; ++d) {
        const uint32_t w = raw[j][d];
        const float lo = (bf2f((u16)(w & 0xFFFFu)) - mu) * rs * gv[2 * d]     + bv[2 * d];
        const float hi = (bf2f((u16)(w >> 16))     - mu) * rs * gv[2 * d + 1] + bv[2 * d + 1];
        o[d] = (uint32_t)f2bf(lo) | ((uint32_t)f2bf(hi) << 16);
      }
      *(vu4*)(rowp + (size_t)c * 2) = o;
    }
  }
  __syncthreads();

  // ---------------- Phase 4: GEMM2 (64x1024 @ 1024x512) + bias + SiLU --------
#pragma unroll 1
  for (int mt = 0; mt < 4; ++mt) {
    const int m0 = mt * 16;
#pragma unroll 1
    for (int np = 0; np < 2; ++np) {
      const int n0 = wave * 64 + np * 32;
      v8f acc0 = {}, acc1 = {};
#pragma unroll 4
      for (int kt = 0; kt < 32; ++kt) {
        const int k0 = kt * 32;
        const v16bf a  = lds_a_frag(H1, m0, k0, lane);
        const v16bf bA = glb_b_frag(W2T, n0,      k0, lane);
        const v16bf bB = glb_b_frag(W2T, n0 + 16, k0, lane);
        acc0 = wmma_bf16(a, bA, acc0);
        acc1 = wmma_bf16(a, bB, acc1);
      }
      const int half = lane >> 4;
      const int nA = n0 + (lane & 15), nB = nA + 16;
      const float biA = b2[nA], biB = b2[nB];
#pragma unroll
      for (int r = 0; r < 8; ++r) {
        const int m = m0 + r + 8 * half;
        const int rg = row0 + m;
        if (rg < N) {
          out[(size_t)rg * FDIM + nA] = silu(acc0[r] + biA);
          out[(size_t)rg * FDIM + nB] = silu(acc1[r] + biB);
        }
      }
    }
  }
}

// ---------------------------------------------------------------------------
// Host launcher
// ---------------------------------------------------------------------------
extern "C" void kernel_launch(void* const* d_in, const int* in_sizes, int n_in,
                              void* d_out, int out_size, void* d_ws, size_t ws_size,
                              hipStream_t stream) {
  // setup_inputs order:
  // 0 coords, 1 nodes, 2 comps, 3 aggr_coords, 4 aggr_nodes, 5 aggr_comps,
  // 6 ln1_g, 7 ln1_b, 8 W1, 9 b1, 10 ln2_g, 11 ln2_b, 12 W2, 13 b2
  const float* nodes      = (const float*)d_in[1];
  const int*   comps      = (const int*)d_in[2];
  const float* aggr_nodes = (const float*)d_in[4];
  const int*   aggr_comps = (const int*)d_in[5];
  const float* ln1_g = (const float*)d_in[6];
  const float* ln1_b = (const float*)d_in[7];
  const float* W1    = (const float*)d_in[8];
  const float* b1    = (const float*)d_in[9];
  const float* ln2_g = (const float*)d_in[10];
  const float* ln2_b = (const float*)d_in[11];
  const float* W2    = (const float*)d_in[12];
  const float* b2    = (const float*)d_in[13];

  const int N = in_sizes[1] / FDIM;     // 100000
  const int M = in_sizes[5];            // 16384

  // Workspace: n-major bf16 weight copies (L2-resident, WMMA-B-friendly)
  u16* W1T = (u16*)d_ws;                          // 1024 x 1024 bf16 (2 MB)
  u16* W2T = W1T + (size_t)TWOF * TWOF;           //  512 x 1024 bf16 (1 MB)

  transpose_f32_to_bf16<<<dim3(TWOF / 32, TWOF / 32), dim3(32, 8), 0, stream>>>(
      W1, W1T, TWOF, TWOF);
  transpose_f32_to_bf16<<<dim3(FDIM / 32, TWOF / 32), dim3(32, 8), 0, stream>>>(
      W2, W2T, TWOF, FDIM);

  const size_t lds_bytes = (size_t)2 * MT * ROWB;  // 258 KB (<320 KB/WGP)
  hipFuncSetAttribute((const void*)fused_mlp_kernel,
                      hipFuncAttributeMaxDynamicSharedMemorySize, (int)lds_bytes);

  const int grid = (N + MT - 1) / MT;
  fused_mlp_kernel<<<grid, 256, lds_bytes, stream>>>(
      nodes, comps, aggr_nodes, aggr_comps,
      ln1_g, ln1_b, W1T, b1, ln2_g, ln2_b, W2T, b2,
      (float*)d_out, N, M);
}
